// AtrousFourWayMamba_61220463837940
// MI455X (gfx1250) — matrix operand order repaired and checked by reading, not compiled, not run on hardware
//
#include <hip/hip_runtime.h>
#include <math.h>

typedef _Float16 half_t;
typedef __attribute__((ext_vector_type(16))) _Float16 v16h;
typedef __attribute__((ext_vector_type(8)))  float    v8f;

#define L_SEQ   2000
#define N_PAD   2016         /* L_SEQ padded to even 16-tile count (126 tiles) */
#define D_MODEL 512
#define D_INNER 1024
#define E2      2048
#define D_STATE 16
#define DT_RANK 32
#define XROWS   80           /* x_proj_w output rows (dt, B, C use 0..63) */
#define SCH     25           /* scan chunks   */
#define SCLEN   80           /* steps / chunk */

// ============================================================================
// Operand packing into CDNA5 WMMA-native per-lane layouts (f32 -> f16),
// zero-padded to Mpad/Npad (even tile counts for the 2x2 macro-tile GEMM).
// A operand of v_wmma_f32_16x16x32_f16, tile (tm, kb):
//   lane<16 : row m = tm*16+lane,    halves j -> K = j    (j<8), j+8  (j>=8)
//   lane>=16: row m = tm*16+lane-16, halves j -> K = j+8  (j<8), j+16 (j>=8)
// ============================================================================
__global__ void afwm_pack_a(const float* __restrict__ A, half_t* __restrict__ out,
                            int M, int K, int Mpad) {
  int id = blockIdx.x * blockDim.x + threadIdx.x;
  int KB = K >> 5;
  int total = (Mpad >> 4) * KB * 512;
  if (id >= total) return;
  int j    = id & 15;
  int lane = (id >> 4) & 31;
  int kb   = (id >> 9) % KB;
  int tm   = (id >> 9) / KB;
  int m    = tm * 16 + (lane & 15);
  int kloc = j + ((j >= 8) ? 8 : 0) + ((lane >= 16) ? 8 : 0);
  int k    = kb * 32 + kloc;
  out[id] = (m < M) ? (half_t)A[(size_t)m * K + k] : (half_t)0.f;
}

// B operand (32x16, K x N), tile (tn, kb):
//   lane&15 = column n, halves j -> K = j (j<8), j+8 (j>=8), +8 for lane>=16.
__global__ void afwm_pack_b(const float* __restrict__ B, half_t* __restrict__ out,
                            int K, int N, int Npad, long strideK, long strideN) {
  int id = blockIdx.x * blockDim.x + threadIdx.x;
  int KB = K >> 5;
  int total = (Npad >> 4) * KB * 512;
  if (id >= total) return;
  int j    = id & 15;
  int lane = (id >> 4) & 31;
  int kb   = (id >> 9) % KB;
  int tn   = (id >> 9) / KB;
  int n    = tn * 16 + (lane & 15);
  int k    = kb * 32 + j + ((j >= 8) ? 8 : 0) + ((lane >= 16) ? 8 : 0);
  out[id] = (n < N) ? (half_t)B[(size_t)k * strideK + (size_t)n * strideN]
                    : (half_t)0.f;
}

// ============================================================================
// WMMA GEMM, 2x2 register macro-tile: one wave computes 32x32 of C.
// 4 independent v_wmma accumulation chains per K-step; 32 B of operand
// traffic per WMMA (vs 64 B for 1x1). Stores guarded to true M x N.
// ============================================================================
__device__ __forceinline__ void afwm_store_tile(float* __restrict__ C,
                                                const float* __restrict__ bias,
                                                const v8f& acc, int rowBase, int colBase,
                                                int lane, int M, int N) {
  int col = colBase + (lane & 15);
  if (col >= N) return;
  int rbase = (lane < 16) ? 0 : 8;
#pragma unroll
  for (int r = 0; r < 8; ++r) {
    int row = rowBase + rbase + r;
    if (row < M) {
      float v = acc[r];
      if (bias) v += bias[row];
      C[(size_t)row * N + col] = v;
    }
  }
}

__global__ void afwm_wmma_gemm(const half_t* __restrict__ Apack,
                               const half_t* __restrict__ Bpack,
                               float* __restrict__ C,
                               const float* __restrict__ bias,
                               int NT2 /* Npad/32 */, int KB, int M, int N) {
  int mac = blockIdx.x;
  int bm = mac / NT2;
  int bn = mac % NT2;
  int lane = threadIdx.x;
  size_t tstride = (size_t)KB * 512;                       // halves per 16-tile
  const half_t* a0p = Apack + (size_t)(2 * bm)     * tstride + lane * 16;
  const half_t* a1p = a0p + tstride;
  const half_t* b0p = Bpack + (size_t)(2 * bn)     * tstride + lane * 16;
  const half_t* b1p = b0p + tstride;
  v8f acc00 = {}, acc01 = {}, acc10 = {}, acc11 = {};
  for (int kb = 0; kb < KB; ++kb) {
    v16h a0 = *(const v16h*)a0p; a0p += 512;
    v16h a1 = *(const v16h*)a1p; a1p += 512;
    v16h b0 = *(const v16h*)b0p; b0p += 512;
    v16h b1 = *(const v16h*)b1p; b1p += 512;
    acc00 = __builtin_amdgcn_wmma_f32_16x16x32_f16(false, a0, false, b0, (short)0, acc00, false, false);
    acc01 = __builtin_amdgcn_wmma_f32_16x16x32_f16(false, a0, false, b1, (short)0, acc01, false, false);
    acc10 = __builtin_amdgcn_wmma_f32_16x16x32_f16(false, a1, false, b0, (short)0, acc10, false, false);
    acc11 = __builtin_amdgcn_wmma_f32_16x16x32_f16(false, a1, false, b1, (short)0, acc11, false, false);
  }
  int r0 = bm * 32, c0 = bn * 32;
  afwm_store_tile(C, bias, acc00, r0,      c0,      lane, M, N);
  afwm_store_tile(C, bias, acc01, r0,      c0 + 16, lane, M, N);
  afwm_store_tile(C, bias, acc10, r0 + 16, c0,      lane, M, N);
  afwm_store_tile(C, bias, acc11, r0 + 16, c0 + 16, lane, M, N);
}

// ============================================================================
// Atrous permutation maps (rate = 10, L = 2000, h = 200 hardcoded).
// ============================================================================
__device__ __forceinline__ int afwm_s2_src(int pos) {
  int seg = pos / 500, p = pos % 500;
  if (seg == 0) { int ih = p / 5,   jh = p % 5;   return 20*ih + 2*jh; }
  if (seg == 1) { int jh = p / 100, ih = p % 100; return 20*ih + 10 + 2*jh; }
  if (seg == 2) { int pp = 499 - p; int ih = pp/5,   jh = pp%5;   return 20*ih + 2*jh + 1; }
  {              int pp = 499 - p; int jh = pp/100, ih = pp%100; return 20*ih + 10 + 2*jh + 1; }
}
__device__ __forceinline__ int afwm_s3_src(int pos) {
  if (pos < 1000) return 20*(pos/10) + pos%10;
  int p = pos - 1000; return 20*(p/10) + 10 + p%10;
}
__device__ __forceinline__ int afwm_s2_pos(int l) {
  int ih = l / 20, t = l % 20;
  if (t < 10) { int jh = t >> 1; return (t & 1) ? (1000 + 499 - (ih*5 + jh)) : (ih*5 + jh); }
  int c = t - 10, jh = c >> 1;
  return (c & 1) ? (1500 + 499 - (jh*100 + ih)) : (500 + jh*100 + ih);
}
__device__ __forceinline__ int afwm_s3_pos(int l) {
  int ih = l / 20, t = l % 20;
  return (t < 10) ? (ih*10 + t) : (1000 + ih*10 + (t - 10));
}

__global__ void afwm_gather(const float* __restrict__ xz, float* __restrict__ xs, int mode) {
  int id = blockIdx.x * blockDim.x + threadIdx.x;
  if (id >= E2 * L_SEQ) return;
  int d = id / L_SEQ, pos = id % L_SEQ;
  int l = mode ? afwm_s3_src(pos) : afwm_s2_src(pos);
  xs[id] = xz[(size_t)d * L_SEQ + l];
}

// ============================================================================
// Depthwise causal conv (k=4) + SiLU on rows 0..1023 of xz.
// ============================================================================
__global__ void afwm_dwconv_silu(const float* __restrict__ xz,
                                 const float* __restrict__ w,
                                 const float* __restrict__ b,
                                 float* __restrict__ u) {
  int id = blockIdx.x * blockDim.x + threadIdx.x;
  if (id >= D_INNER * L_SEQ) return;
  int d = id / L_SEQ, l = id % L_SEQ;
  const float* xr = xz + (size_t)d * L_SEQ;
  float acc = b[d];
#pragma unroll
  for (int t = 0; t < 4; ++t) {
    int ll = l - 3 + t;
    if (ll >= 0) acc += w[d*4 + t] * xr[ll];
  }
  u[id] = acc / (1.f + __expf(-acc));
}

__global__ void afwm_softplus_bias(float* __restrict__ delta,
                                   const float* __restrict__ bias) {
  int id = blockIdx.x * blockDim.x + threadIdx.x;
  if (id >= D_INNER * L_SEQ) return;
  int d = id / L_SEQ;
  float x = delta[id] + bias[d];
  delta[id] = (x > 20.f) ? x : log1pf(__expf(x));
}

// ============================================================================
// Chunked parallel selective scan (h <- a*h + b is associative).
// Phase 1: per chunk, P = prod(a), Q = h run from h0=0.     depth SCLEN
// Phase 2: serial prefix over SCH chunks -> incoming state. depth SCH
// Phase 3: replay chunk from true h0, emit gated output.    depth SCLEN
// 16 lanes per channel carry D_STATE; sum_n h*C via width-16 __shfl_xor.
// ============================================================================
__global__ void afwm_scan_ph1(const float* __restrict__ u, const float* __restrict__ delta,
                              const float* __restrict__ xdbl, const float* __restrict__ A_log,
                              float* __restrict__ P, float* __restrict__ Q) {
  int tid = threadIdx.x;                  // 512 = 32 channels x 16 states
  int n = tid & 15;
  int dgrp  = blockIdx.x / SCH;
  int chunk = blockIdx.x % SCH;
  int d = dgrp * 32 + (tid >> 4);
  float A = -__expf(A_log[d * D_STATE + n]);
  const float* Brow = xdbl + (size_t)(DT_RANK + n) * L_SEQ;
  const float* urow = u     + (size_t)d * L_SEQ;
  const float* drow = delta + (size_t)d * L_SEQ;
  int l0 = chunk * SCLEN;
  __builtin_prefetch(drow + l0, 0, 1);
  __builtin_prefetch(urow + l0, 0, 1);
  __builtin_prefetch(Brow + l0, 0, 1);
  float h = 0.f, p = 1.f;
  for (int i = 0; i < SCLEN; ++i) {
    int l = l0 + i;
    float dl = drow[l], ul = urow[l];
    float a = __expf(dl * A);
    h = a * h + dl * ul * Brow[l];
    p *= a;
  }
  size_t idx = ((size_t)d * D_STATE + n) * SCH + chunk;
  P[idx] = p; Q[idx] = h;
}

__global__ void afwm_scan_ph2(const float* __restrict__ P, const float* __restrict__ Q,
                              float* __restrict__ Hin) {
  int id = blockIdx.x * blockDim.x + threadIdx.x;   // D_INNER * D_STATE
  if (id >= D_INNER * D_STATE) return;
  size_t base = (size_t)id * SCH;
  float h = 0.f;
  for (int c = 0; c < SCH; ++c) {
    Hin[base + c] = h;
    h = P[base + c] * h + Q[base + c];
  }
}

__global__ void afwm_scan_ph3(const float* __restrict__ u, const float* __restrict__ delta,
                              const float* __restrict__ xdbl, const float* __restrict__ A_log,
                              const float* __restrict__ Dvec, const float* __restrict__ xzperm,
                              const float* __restrict__ Hin, float* __restrict__ y) {
  int tid = threadIdx.x;
  int n = tid & 15;
  int dgrp  = blockIdx.x / SCH;
  int chunk = blockIdx.x % SCH;
  int d = dgrp * 32 + (tid >> 4);
  float A  = -__expf(A_log[d * D_STATE + n]);
  float Dd = Dvec[d];
  const float* Brow = xdbl + (size_t)(DT_RANK + n) * L_SEQ;
  const float* Crow = xdbl + (size_t)(DT_RANK + D_STATE + n) * L_SEQ;
  const float* urow = u     + (size_t)d * L_SEQ;
  const float* drow = delta + (size_t)d * L_SEQ;
  const float* zrow = xzperm + (size_t)(D_INNER + d) * L_SEQ;
  float* yrow = y + (size_t)d * L_SEQ;
  int l0 = chunk * SCLEN;
  __builtin_prefetch(drow + l0, 0, 1);
  __builtin_prefetch(urow + l0, 0, 1);
  __builtin_prefetch(Crow + l0, 0, 1);
  float h = Hin[((size_t)d * D_STATE + n) * SCH + chunk];
  for (int i = 0; i < SCLEN; ++i) {
    int l = l0 + i;
    float dl = drow[l], ul = urow[l];
    h = __expf(dl * A) * h + dl * ul * Brow[l];
    float part = h * Crow[l];
    for (int off = 8; off; off >>= 1) part += __shfl_xor(part, off, 16);
    if (n == 0) {
      float z = zrow[l];
      yrow[l] = (part + Dd * ul) * (z / (1.f + __expf(-z)));
    }
  }
}

// out[d,l] = o_s2[d, s2pos(l)] + o_s3[d, s3pos(l)] + o_c[d, l]
__global__ void afwm_combine(const float* __restrict__ o0, const float* __restrict__ o1,
                             const float* __restrict__ o2, float* __restrict__ out) {
  int id = blockIdx.x * blockDim.x + threadIdx.x;
  if (id >= D_INNER * L_SEQ) return;
  int d = id / L_SEQ, l = id % L_SEQ;
  size_t base = (size_t)d * L_SEQ;
  out[id] = o0[base + afwm_s2_pos(l)] + o1[base + afwm_s3_pos(l)] + o2[id];
}

// ============================================================================
// bi_attn small ops
// ============================================================================
__global__ void afwm_ln_stats(const float* __restrict__ out,
                              float* __restrict__ mu, float* __restrict__ inv) {
  int l = blockIdx.x;
  __shared__ float s1[256], s2[256];
  float a = 0.f, b = 0.f;
  for (int d = threadIdx.x; d < D_INNER; d += 256) {
    float v = out[(size_t)d * L_SEQ + l];
    a += v; b += v * v;
  }
  s1[threadIdx.x] = a; s2[threadIdx.x] = b;
  __syncthreads();
  for (int s = 128; s; s >>= 1) {
    if (threadIdx.x < s) { s1[threadIdx.x] += s1[threadIdx.x + s];
                           s2[threadIdx.x] += s2[threadIdx.x + s]; }
    __syncthreads();
  }
  if (threadIdx.x == 0) {
    float m = s1[0] * (1.f / D_INNER);
    float var = s2[0] * (1.f / D_INNER) - m * m;
    mu[l] = m; inv[l] = rsqrtf(var + 1e-5f);
  }
}

__global__ void afwm_mean_xn(const float* __restrict__ out, const float* __restrict__ mu,
                             const float* __restrict__ inv, const float* __restrict__ ln_g,
                             const float* __restrict__ ln_b, float* __restrict__ g) {
  int d = blockIdx.x * blockDim.x + threadIdx.x;
  if (d >= D_INNER) return;
  float s = 0.f;
  for (int l = 0; l < L_SEQ; ++l) s += (out[(size_t)d * L_SEQ + l] - mu[l]) * inv[l];
  g[d] = ln_b[d] + ln_g[d] * (s * (1.f / L_SEQ));
}

__global__ void afwm_gr_gelu(const float* __restrict__ g, const float* __restrict__ w,
                             const float* __restrict__ b, float* __restrict__ g2) {
  int j = blockIdx.x * blockDim.x + threadIdx.x;
  if (j >= 512) return;
  float s = b[j];
  for (int d = 0; d < D_INNER; ++d) s += g[d] * w[j * D_INNER + d];
  g2[j] = 0.5f * s * (1.f + erff(s * 0.70710678118654752f));
}

__global__ void afwm_cs_sigmoid(const float* __restrict__ g2, const float* __restrict__ w,
                                const float* __restrict__ b, float* __restrict__ attn) {
  int i = blockIdx.x * blockDim.x + threadIdx.x;
  if (i >= D_INNER) return;
  float s = b[i];
  for (int j = 0; j < 512; ++j) s += g2[j] * w[i * 512 + j];
  attn[i] = 1.f / (1.f + __expf(-s));
}

__global__ void afwm_scale(const float* __restrict__ out, const float* __restrict__ attn,
                           float* __restrict__ scaled) {
  int id = blockIdx.x * blockDim.x + threadIdx.x;
  if (id >= D_INNER * L_SEQ) return;
  scaled[id] = out[id] * attn[id / L_SEQ];
}

// final[m,l] (512 x 2000) -> d_out[l,m]
__global__ void afwm_store_out(const float* __restrict__ fin, float* __restrict__ out) {
  int id = blockIdx.x * blockDim.x + threadIdx.x;
  if (id >= D_MODEL * L_SEQ) return;
  int m = id % D_MODEL, l = id / D_MODEL;
  out[id] = fin[(size_t)m * L_SEQ + l];
}

// ============================================================================
// Host orchestration
// ============================================================================
extern "C" void kernel_launch(void* const* d_in, const int* in_sizes, int n_in,
                              void* d_out, int out_size, void* d_ws, size_t ws_size,
                              hipStream_t stream) {
  (void)in_sizes; (void)n_in; (void)out_size;

  const float* hidden     = (const float*)d_in[0];
  const float* in_proj_w  = (const float*)d_in[1];
  const float* conv_w[3]  = {(const float*)d_in[2],  (const float*)d_in[4],  (const float*)d_in[6]};
  const float* conv_b[3]  = {(const float*)d_in[3],  (const float*)d_in[5],  (const float*)d_in[7]};
  const float* xproj_w[3] = {(const float*)d_in[8],  (const float*)d_in[9],  (const float*)d_in[10]};
  const float* dtprj_w[3] = {(const float*)d_in[11], (const float*)d_in[12], (const float*)d_in[13]};
  const float* dt_bias[3] = {(const float*)d_in[14], (const float*)d_in[15], (const float*)d_in[16]};
  const float* A_log[3]   = {(const float*)d_in[17], (const float*)d_in[18], (const float*)d_in[19]};
  const float* Dvec[3]    = {(const float*)d_in[20], (const float*)d_in[21], (const float*)d_in[22]};
  const float* out_proj_w = (const float*)d_in[23];
  const float* ln_g = (const float*)d_in[24];
  const float* ln_b = (const float*)d_in[25];
  const float* gr_w = (const float*)d_in[26];
  const float* gr_b = (const float*)d_in[27];
  const float* cs_w = (const float*)d_in[28];
  const float* cs_b = (const float*)d_in[29];
  const float* ow   = (const float*)d_in[30];
  const float* ob   = (const float*)d_in[31];
  // d_in[32] = rate (==10; permutation maps are specialized to rate=10, L=2000)

  // ---- workspace arena (floats), ~85.6 MB total ----
  const size_t OFF_XZ    = 0;
  const size_t OFF_XS    = OFF_XZ    + (size_t)E2 * L_SEQ;       // later OUT + FINAL
  const size_t OFF_U     = OFF_XS    + (size_t)E2 * L_SEQ;       // later scaled
  const size_t OFF_DELTA = OFF_U     + (size_t)D_INNER * L_SEQ;  // later attn-out
  const size_t OFF_XDBL  = OFF_DELTA + (size_t)D_INNER * L_SEQ;
  const size_t OFF_O0    = OFF_XDBL  + (size_t)XROWS * L_SEQ;
  const size_t OFF_O1    = OFF_O0    + (size_t)D_INNER * L_SEQ;
  const size_t OFF_O2    = OFF_O1    + (size_t)D_INNER * L_SEQ;
  const size_t OFF_STATS = OFF_O2    + (size_t)D_INNER * L_SEQ;
  const size_t OFF_P     = OFF_STATS + 8192;
  const size_t OFF_Q     = OFF_P     + (size_t)D_INNER * D_STATE * SCH;
  const size_t OFF_HIN   = OFF_Q     + (size_t)D_INNER * D_STATE * SCH;
  const size_t FLT_END   = OFF_HIN   + (size_t)D_INNER * D_STATE * SCH;
  const size_t HB_A      = FLT_END * 4;                          // A pack, 2 MB
  const size_t HB_B      = HB_A + (size_t)2 * 1024 * 1024;       // B pack, 4 MB
  const size_t WS_NEED   = HB_B + (size_t)4 * 1024 * 1024;
  if (ws_size < WS_NEED) return;

  float* W      = (float*)d_ws;
  float* XZ     = W + OFF_XZ;
  float* XS     = W + OFF_XS;
  float* U      = W + OFF_U;
  float* DELTA  = W + OFF_DELTA;
  float* XDBL   = W + OFF_XDBL;
  float* O[3]   = {W + OFF_O0, W + OFF_O1, W + OFF_O2};
  float* MU     = W + OFF_STATS;
  float* INV    = MU + L_SEQ;
  float* G      = INV + L_SEQ;
  float* G2     = G + D_INNER;
  float* ATT    = G2 + 512;
  float* SP     = W + OFF_P;
  float* SQ     = W + OFF_Q;
  float* HIN    = W + OFF_HIN;
  float* OUT    = XS;                         // alias: free after scans
  float* FINAL  = XS + (size_t)D_INNER * L_SEQ;
  float* ATTOUT = DELTA;                      // alias: free after scans
  half_t* AP = (half_t*)((char*)d_ws + HB_A);
  half_t* BP = (half_t*)((char*)d_ws + HB_B);
  float* out = (float*)d_out;

  const int TB  = 256;
  const int NT2 = N_PAD / 32;                 // 63 macro columns
  #define GRID1(nelem) dim3(((nelem) + TB - 1) / TB), dim3(TB), 0, stream

  // ---- G1: xz = in_proj_w (2048x512) x hidden^T (512x2000) ----
  afwm_pack_a<<<GRID1(128 * 16 * 512)>>>(in_proj_w, AP, E2, D_MODEL, 2048);
  afwm_pack_b<<<GRID1(126 * 16 * 512)>>>(hidden, BP, D_MODEL, L_SEQ, N_PAD, 1L, (long)D_MODEL);
  afwm_wmma_gemm<<<dim3(64 * NT2), dim3(32), 0, stream>>>(AP, BP, XZ, nullptr, NT2, 16, E2, L_SEQ);

  // ---- three mamba_inner variants, sequential, reusing scratch ----
  for (int v = 0; v < 3; ++v) {
    const float* src;
    if (v < 2) {
      afwm_gather<<<GRID1(E2 * L_SEQ)>>>(XZ, XS, v);   // v=0 -> s2, v=1 -> s3
      src = XS;
    } else {
      src = XZ;                                        // variant c: identity
    }
    afwm_dwconv_silu<<<GRID1(D_INNER * L_SEQ)>>>(src, conv_w[v], conv_b[v], U);

    // x_dbl = x_proj_w (80x1024) x U (1024x2000)  [M padded 80 -> 96]
    afwm_pack_a<<<GRID1(6 * 32 * 512)>>>(xproj_w[v], AP, XROWS, D_INNER, 96);
    afwm_pack_b<<<GRID1(126 * 32 * 512)>>>(U, BP, D_INNER, L_SEQ, N_PAD, (long)L_SEQ, 1L);
    afwm_wmma_gemm<<<dim3(3 * NT2), dim3(32), 0, stream>>>(AP, BP, XDBL, nullptr, NT2, 32, XROWS, L_SEQ);

    // delta = softplus(dt_proj_w (1024x32) x x_dbl[0:32] + dt_bias)
    afwm_pack_a<<<GRID1(64 * 1 * 512)>>>(dtprj_w[v], AP, D_INNER, DT_RANK, 1024);
    afwm_pack_b<<<GRID1(126 * 1 * 512)>>>(XDBL, BP, DT_RANK, L_SEQ, N_PAD, (long)L_SEQ, 1L);
    afwm_wmma_gemm<<<dim3(32 * NT2), dim3(32), 0, stream>>>(AP, BP, DELTA, nullptr, NT2, 1, D_INNER, L_SEQ);
    afwm_softplus_bias<<<GRID1(D_INNER * L_SEQ)>>>(DELTA, dt_bias[v]);

    // chunked parallel selective scan + gate
    afwm_scan_ph1<<<dim3((D_INNER / 32) * SCH), dim3(512), 0, stream>>>(U, DELTA, XDBL, A_log[v], SP, SQ);
    afwm_scan_ph2<<<GRID1(D_INNER * D_STATE)>>>(SP, SQ, HIN);
    afwm_scan_ph3<<<dim3((D_INNER / 32) * SCH), dim3(512), 0, stream>>>(U, DELTA, XDBL, A_log[v],
                                                                        Dvec[v], src, HIN, O[v]);
  }

  // ---- restore + sum the three branches ----
  afwm_combine<<<GRID1(D_INNER * L_SEQ)>>>(O[0], O[1], O[2], OUT);

  // ---- bi_attn ----
  afwm_ln_stats<<<dim3(L_SEQ), dim3(256), 0, stream>>>(OUT, MU, INV);
  afwm_mean_xn<<<GRID1(D_INNER)>>>(OUT, MU, INV, ln_g, ln_b, G);
  afwm_gr_gelu<<<GRID1(512)>>>(G, gr_w, gr_b, G2);
  afwm_cs_sigmoid<<<GRID1(D_INNER)>>>(G2, cs_w, cs_b, ATT);
  afwm_scale<<<GRID1(D_INNER * L_SEQ)>>>(OUT, ATT, U);   // scaled -> U

  // attn-out GEMM: att_out_w (1024x1024) x scaled + ob
  afwm_pack_a<<<GRID1(64 * 32 * 512)>>>(ow, AP, D_INNER, D_INNER, 1024);
  afwm_pack_b<<<GRID1(126 * 32 * 512)>>>(U, BP, D_INNER, L_SEQ, N_PAD, (long)L_SEQ, 1L);
  afwm_wmma_gemm<<<dim3(32 * NT2), dim3(32), 0, stream>>>(AP, BP, ATTOUT, ob, NT2, 32, D_INNER, L_SEQ);

  // out_proj GEMM: out_proj_w (512x1024) x attn-out
  afwm_pack_a<<<GRID1(32 * 32 * 512)>>>(out_proj_w, AP, D_MODEL, D_INNER, 512);
  afwm_pack_b<<<GRID1(126 * 32 * 512)>>>(ATTOUT, BP, D_INNER, L_SEQ, N_PAD, (long)L_SEQ, 1L);
  afwm_wmma_gemm<<<dim3(16 * NT2), dim3(32), 0, stream>>>(AP, BP, FINAL, nullptr, NT2, 32, D_MODEL, L_SEQ);

  // transpose to (B, L, D_MODEL)
  afwm_store_out<<<GRID1(D_MODEL * L_SEQ)>>>(FINAL, out);
  #undef GRID1
}
